// RelativeMultiheadAttention_71734543778331
// MI455X (gfx1250) — compile-verified, matrix operand order checked by reading, not run on hardware
//
#include <hip/hip_runtime.h>
#include <hip/hip_bf16.h>

// ---------------------------------------------------------------------------
// Transformer-XL relative multi-head attention, fused flash-style, bf16 WMMA.
// B=2, T=2048, DIM=512, H=8, Dh=64.  All matrix math on v_wmma_f32_16x16x32_bf16.
//
// Pipeline:
//   prep:  x,r -> bf16 ; W_qkv,W_r,W_out -> bf16 transposed [N][K]
//   1) qkv projection (WMMA GEMM), folding u/v into q; V stored transposed
//   2) r projection (WMMA GEMM)
//   3) fused causal flash attention with rel-shift band (WMMA + LDS gather)
//   4) output projection + bias (WMMA GEMM)
// Every WMMA operand is fetched as two 16B contiguous runs per lane, matching
// the CDNA5 16-bit operand layout (K = half*8+[0..7] and 16+half*8+[0..7]).
// ---------------------------------------------------------------------------

#define B_   2
#define T_   2048
#define DIM_ 512
#define H_   8
#define DH_  64

typedef __attribute__((ext_vector_type(16))) __bf16 v16bf;
typedef __attribute__((ext_vector_type(8)))  __bf16 v8bf;
typedef __attribute__((ext_vector_type(4)))  __bf16 v4bf;
typedef __attribute__((ext_vector_type(8)))  float  v8f;

static __device__ __forceinline__ v8f wmma_bf16(v16bf a, v16bf b, v8f c) {
  // (neg_a, A, neg_b, B, c_mod, C, reuse_a, reuse_b)
  return __builtin_amdgcn_wmma_f32_16x16x32_bf16(false, a, false, b, (short)0, c,
                                                 false, false);
}

// Assemble a 16x32 16-bit operand from a per-lane row pointer: two contiguous
// 8-element (16B) runs at [half*8] and [16 + half*8].
static __device__ __forceinline__ v16bf load_runs_bf16(const __bf16* row,
                                                       int half) {
  const v8bf lo = *(const v8bf*)(row + half * 8);
  const v8bf hi = *(const v8bf*)(row + 16 + half * 8);
  return __builtin_shufflevector(lo, hi, 0, 1, 2, 3, 4, 5, 6, 7,
                                 8, 9, 10, 11, 12, 13, 14, 15);
}

// A tile 16(M) x 32(K), row-major bf16 (K contiguous).  Also serves as the
// B^T operand (rows indexed by N-lane, contiguous K): m and n are both lane&15.
static __device__ __forceinline__ v16bf load_a_bf16(const __bf16* p, int ld) {
  int lane = threadIdx.x & 31, half = lane >> 4, m = lane & 15;
  return load_runs_bf16(p + (size_t)m * ld, half);
}

// B operand for the BD band: B[k][n] = r_h[T-1-(dbase+n)][h*DH + k0 + k].
// Row depends on the N lane, K contiguous; clamped rows are causally masked.
static __device__ __forceinline__ v16bf load_b_rrev(const __bf16* rh, int h,
                                                    int dbase, int k0) {
  int lane = threadIdx.x & 31, half = lane >> 4, n = lane & 15;
  int d  = dbase + n;
  int rr = (T_ - 1) - d;
  rr = rr < 0 ? 0 : (rr > T_ - 1 ? T_ - 1 : rr);
  return load_runs_bf16(rh + (size_t)rr * DIM_ + h * DH_ + k0, half);
}

// ---------------------------------------------------------------------------
// Prep kernel A: f32 -> bf16, vectorized (b128 in, b64 out).
// ---------------------------------------------------------------------------
__global__ __launch_bounds__(256) void cvt_bf16_kernel(
    const float* __restrict__ src, __bf16* __restrict__ dst, int nelem) {
  int i = (blockIdx.x * 256 + threadIdx.x) * 4;
  if (i < nelem) {
    float4 f = *(const float4*)(src + i);
    v4bf o;
    o[0] = (__bf16)f.x; o[1] = (__bf16)f.y;
    o[2] = (__bf16)f.z; o[3] = (__bf16)f.w;
    *(v4bf*)(dst + i) = o;
  }
}

// ---------------------------------------------------------------------------
// Prep kernel B: weight transpose + convert.  src is (512 x N) f32 row-major,
// dst is (N x 512) bf16 row-major.  Output-coalesced single pass.
// ---------------------------------------------------------------------------
__global__ __launch_bounds__(256) void transpose_w_kernel(
    const float* __restrict__ src, __bf16* __restrict__ dst, int N) {
  int idx = blockIdx.x * 256 + threadIdx.x;   // idx = n*512 + k
  if (idx < N * DIM_) {
    int k = idx & (DIM_ - 1);
    int n = idx >> 9;
    dst[idx] = (__bf16)src[(size_t)k * N + n];
  }
}

// ---------------------------------------------------------------------------
// Kernel 1: qkv = x @ W_qkv  (4096 x 1536 x 512), bf16 operands.
// blockIdx.y selects q/k/v -> scalar epilogue branch.  V stored transposed
// (vvT[b][d][s]) so attention's P@V B-operand loads are contiguous-K b128.
// ---------------------------------------------------------------------------
__global__ __launch_bounds__(128) void qkv_proj_kernel(
    const __bf16* __restrict__ xbf, const __bf16* __restrict__ WqkvT,
    const float* __restrict__ uvec, const float* __restrict__ vvec,
    __bf16* __restrict__ qu, __bf16* __restrict__ qvo,
    __bf16* __restrict__ kk, __bf16* __restrict__ vvT) {
  const int NT = DIM_ / 16;  // 32 N-tiles per selection
  int sel = blockIdx.y;      // 0=q, 1=k, 2=v  (scalar)
  int gw = blockIdx.x * 4 + (threadIdx.x >> 5);
  int m0 = (gw / NT) * 16, n0 = (gw % NT) * 16;  // n0: column within 512
  const __bf16* wt = WqkvT + ((size_t)sel * DIM_ + n0) * DIM_;
  v8f c = {};
  for (int k0 = 0; k0 < DIM_; k0 += 32) {
    c = wmma_bf16(load_a_bf16(xbf + (size_t)m0 * DIM_ + k0, DIM_),
                  load_a_bf16(wt + k0, DIM_), c);
  }
  int lane = threadIdx.x & 31, half = lane >> 4, n = lane & 15;
  int hc = n0 + n;  // h*64 + d
  if (sel == 0) {
    float ub = uvec[hc], vb = vvec[hc];
#pragma unroll
    for (int i = 0; i < 8; ++i) {
      size_t o = ((size_t)m0 + i + 8 * half) * DIM_ + hc;
      qu[o]  = (__bf16)(c[i] + ub);   // fold u into q
      qvo[o] = (__bf16)(c[i] + vb);   // fold v into q
    }
  } else if (sel == 1) {
#pragma unroll
    for (int i = 0; i < 8; ++i)
      kk[((size_t)m0 + i + 8 * half) * DIM_ + hc] = (__bf16)c[i];
  } else {
#pragma unroll
    for (int i = 0; i < 8; ++i) {
      int row = m0 + i + 8 * half;        // b*T + t
      int b = row >> 11, t = row & (T_ - 1);
      vvT[((size_t)b * DIM_ + hc) * T_ + t] = (__bf16)c[i];
    }
  }
}

// ---------------------------------------------------------------------------
// Kernel 2: r_h = r @ W_r  (2048 x 512 x 512), bf16 operands + out.
// ---------------------------------------------------------------------------
__global__ __launch_bounds__(128) void rel_proj_kernel(
    const __bf16* __restrict__ rbf, const __bf16* __restrict__ WrT,
    __bf16* __restrict__ rh) {
  const int NT = DIM_ / 16;  // 32
  int gw = blockIdx.x * 4 + (threadIdx.x >> 5);
  int m0 = (gw / NT) * 16, n0 = (gw % NT) * 16;
  v8f c = {};
  for (int k0 = 0; k0 < DIM_; k0 += 32) {
    c = wmma_bf16(load_a_bf16(rbf + (size_t)m0 * DIM_ + k0, DIM_),
                  load_a_bf16(WrT + (size_t)n0 * DIM_ + k0, DIM_), c);
  }
  int lane = threadIdx.x & 31, half = lane >> 4, n = lane & 15;
#pragma unroll
  for (int i = 0; i < 8; ++i)
    rh[((size_t)m0 + i + 8 * half) * DIM_ + n0 + n] = (__bf16)c[i];
}

// ---------------------------------------------------------------------------
// Kernel 3: fused causal relative attention (flash-style).
// One wave owns a 16-row query tile of one (b,h); streams 32-key blocks.
//   score = [ (q+u)·k  +  (q+v)·r_rev[t-s] ] * Dh^-0.5,   r_rev[d]=r_h[T-1-d]
// BD computed as a 16x48 WMMA band over relative offsets, gathered via LDS.
// ---------------------------------------------------------------------------
__global__ __launch_bounds__(128) void relattn_kernel(
    const __bf16* __restrict__ qu, const __bf16* __restrict__ qvb,
    const __bf16* __restrict__ kk, const __bf16* __restrict__ vvT,
    const __bf16* __restrict__ rh, __bf16* __restrict__ ao) {
  __shared__ __align__(16) char smem[4 * 4096];
  int wave = threadIdx.x >> 5;
  int lane = threadIdx.x & 31, half = lane >> 4, n = lane & 15;
  float*  bdls = (float*)(smem + wave * 4096);          // 16 x 48 f32 band
  __bf16* pls  = (__bf16*)(smem + wave * 4096 + 3072);  // 16 x 32 bf16 probs

  const int QT = T_ / 16;  // 128 query tiles per (b,h)
  int tile = blockIdx.x * 4 + wave;
  int t0 = (tile % QT) * 16;
  int bh = tile / QT;
  int h = bh % H_, b = bh / H_;
  const size_t bt = (size_t)b * T_;

  const __bf16* qup = qu  + (bt + t0) * DIM_ + h * DH_;
  const __bf16* qvp = qvb + (bt + t0) * DIM_ + h * DH_;
  v16bf aqu0 = load_a_bf16(qup, DIM_), aqu1 = load_a_bf16(qup + 32, DIM_);
  v16bf aqv0 = load_a_bf16(qvp, DIM_), aqv1 = load_a_bf16(qvp + 32, DIM_);

  float m_i[8], l_i[8];
  v8f o0 = {}, o1 = {}, o2 = {}, o3 = {};
#pragma unroll
  for (int i = 0; i < 8; ++i) { m_i[i] = -1e30f; l_i[i] = 0.0f; }

  for (int s0 = 0; s0 <= t0 + 15; s0 += 32) {
    // ---- BD band: band[ti,dc] = (q+v)[t0+ti] . r_rev[dbase+dc], dc in [0,48)
    int dbase = t0 - s0 - 31;
#pragma unroll
    for (int dt = 0; dt < 3; ++dt) {
      v8f c = {};
      c = wmma_bf16(aqv0, load_b_rrev(rh, h, dbase + dt * 16, 0), c);
      c = wmma_bf16(aqv1, load_b_rrev(rh, h, dbase + dt * 16, 32), c);
#pragma unroll
      for (int i = 0; i < 8; ++i)
        bdls[(i + 8 * half) * 48 + dt * 16 + n] = c[i];
    }
    asm volatile("s_wait_dscnt 0x0" ::: "memory");

    // ---- AC (Q·K^T) + gathered BD, causal mask, scale ----
    float sc[2][8];
#pragma unroll
    for (int ns = 0; ns < 2; ++ns) {
      const __bf16* kp = kk + (bt + s0 + ns * 16) * DIM_ + h * DH_;
      v8f c = {};
      c = wmma_bf16(aqu0, load_a_bf16(kp, DIM_), c);       // K rows act as B^T
      c = wmma_bf16(aqu1, load_a_bf16(kp + 32, DIM_), c);
#pragma unroll
      for (int i = 0; i < 8; ++i) {
        int ti = i + 8 * half;
        int sj = ns * 16 + n;
        float s = (c[i] + bdls[ti * 48 + (ti - sj + 31)]) * 0.125f;
        sc[ns][i] = ((s0 + sj) > (t0 + ti)) ? -1e30f : s;
      }
    }

    // ---- online softmax (row stats via 16-lane butterflies, wave32) ----
#pragma unroll
    for (int i = 0; i < 8; ++i) {
      float mx = fmaxf(sc[0][i], sc[1][i]);
#pragma unroll
      for (int d = 1; d < 16; d <<= 1) mx = fmaxf(mx, __shfl_xor(mx, d, 16));
      float mn   = fmaxf(m_i[i], mx);
      float corr = __expf(m_i[i] - mn);
      float p0 = __expf(sc[0][i] - mn);
      float p1 = __expf(sc[1][i] - mn);
      sc[0][i] = p0; sc[1][i] = p1;
      float rs = p0 + p1;
#pragma unroll
      for (int d = 1; d < 16; d <<= 1) rs += __shfl_xor(rs, d, 16);
      l_i[i] = l_i[i] * corr + rs;
      m_i[i] = mn;
      o0[i] *= corr; o1[i] *= corr; o2[i] *= corr; o3[i] *= corr;
    }

    // ---- restage P through LDS into A-operand layout ----
#pragma unroll
    for (int ns = 0; ns < 2; ++ns)
#pragma unroll
      for (int i = 0; i < 8; ++i)
        pls[(i + 8 * half) * 32 + ns * 16 + n] = (__bf16)sc[ns][i];
    asm volatile("s_wait_dscnt 0x0" ::: "memory");
    v16bf pa = load_a_bf16(pls, 32);   // ds_load_b128 x2

    // ---- O += P @ V  (V transposed: row = d-lane, contiguous s -> b128) ----
    const __bf16* vp = vvT + ((size_t)b * DIM_ + h * DH_) * T_ + s0;
    o0 = wmma_bf16(pa, load_a_bf16(vp + (size_t)0  * T_, T_), o0);
    o1 = wmma_bf16(pa, load_a_bf16(vp + (size_t)16 * T_, T_), o1);
    o2 = wmma_bf16(pa, load_a_bf16(vp + (size_t)32 * T_, T_), o2);
    o3 = wmma_bf16(pa, load_a_bf16(vp + (size_t)48 * T_, T_), o3);

    if (s0 + 32 <= t0 + 15) {  // prefetch next K block (uniform branch)
      __builtin_prefetch(kk + (bt + s0 + 32) * DIM_ + h * DH_, 0, 1);
      __builtin_prefetch(vvT + ((size_t)b * DIM_ + h * DH_) * T_ + s0 + 32, 0, 1);
    }
  }

  // ---- normalize and store bf16 per-head output ----
  __bf16* op = ao + (bt + t0) * DIM_ + h * DH_;
#pragma unroll
  for (int i = 0; i < 8; ++i) {
    float inv = 1.0f / l_i[i];
    size_t ro = (size_t)(i + 8 * half) * DIM_;
    op[ro + n + 0]  = (__bf16)(o0[i] * inv);
    op[ro + n + 16] = (__bf16)(o1[i] * inv);
    op[ro + n + 32] = (__bf16)(o2[i] * inv);
    op[ro + n + 48] = (__bf16)(o3[i] * inv);
  }
}

// ---------------------------------------------------------------------------
// Kernel 4: out = attn_out @ W_out + b_out  (4096 x 512 x 512), f32 out.
// ---------------------------------------------------------------------------
__global__ __launch_bounds__(128) void out_proj_kernel(
    const __bf16* __restrict__ ao, const __bf16* __restrict__ WoutT,
    const float* __restrict__ bout, float* __restrict__ out) {
  const int NT = DIM_ / 16;
  int gw = blockIdx.x * 4 + (threadIdx.x >> 5);
  int m0 = (gw / NT) * 16, n0 = (gw % NT) * 16;
  v8f c = {};
  for (int k0 = 0; k0 < DIM_; k0 += 32) {
    c = wmma_bf16(load_a_bf16(ao + (size_t)m0 * DIM_ + k0, DIM_),
                  load_a_bf16(WoutT + (size_t)n0 * DIM_ + k0, DIM_), c);
  }
  int lane = threadIdx.x & 31, half = lane >> 4, n = lane & 15;
  float bias = bout[n0 + n];
#pragma unroll
  for (int i = 0; i < 8; ++i)
    out[((size_t)m0 + i + 8 * half) * DIM_ + n0 + n] = c[i] + bias;
}

// ---------------------------------------------------------------------------
extern "C" void kernel_launch(void* const* d_in, const int* in_sizes, int n_in,
                              void* d_out, int out_size, void* d_ws, size_t ws_size,
                              hipStream_t stream) {
  const float* x    = (const float*)d_in[0];
  const float* r    = (const float*)d_in[1];
  const float* Wqkv = (const float*)d_in[2];
  const float* Wr   = (const float*)d_in[3];
  const float* u    = (const float*)d_in[4];
  const float* v    = (const float*)d_in[5];
  const float* Wout = (const float*)d_in[6];
  const float* bout = (const float*)d_in[7];
  float* out = (float*)d_out;

  // Workspace layout (bf16 buffers), ~30.5 MiB total:
  char* ws = (char*)d_ws;
  __bf16* qu    = (__bf16*)(ws);                         // (B*T, 512)  q + u
  __bf16* qvb   = (__bf16*)(ws + ((size_t)4  << 20));    // (B*T, 512)  q + v
  __bf16* kk    = (__bf16*)(ws + ((size_t)8  << 20));    // (B*T, 512)  K
  __bf16* vvT   = (__bf16*)(ws + ((size_t)12 << 20));    // (B, 512, T) V^T
  __bf16* rh    = (__bf16*)(ws + ((size_t)16 << 20));    // (T, 512)    r @ W_r
  __bf16* ao    = (__bf16*)(ws + ((size_t)18 << 20));    // (B*T, 512)  attn out
  __bf16* xbf   = (__bf16*)(ws + ((size_t)22 << 20));    // (B*T, 512)  x bf16
  __bf16* rbf   = (__bf16*)(ws + ((size_t)26 << 20));    // (T, 512)    r bf16
  __bf16* WqkvT = (__bf16*)(ws + ((size_t)28 << 20));    // (1536, 512) W_qkv^T
  __bf16* WrT   = (__bf16*)(ws + ((size_t)30208 << 10)); // (512, 512)  W_r^T
  __bf16* WoutT = (__bf16*)(ws + ((size_t)30720 << 10)); // (512, 512)  W_out^T

  // ---- prep: convert activations, transpose+convert weights (one pass) ----
  const int nx = B_ * T_ * DIM_;       // 2M
  const int nr = T_ * DIM_;            // 1M
  cvt_bf16_kernel<<<dim3(nx / 1024), 256, 0, stream>>>(x, xbf, nx);
  cvt_bf16_kernel<<<dim3(nr / 1024), 256, 0, stream>>>(r, rbf, nr);
  transpose_w_kernel<<<dim3((3 * DIM_ * DIM_) / 256), 256, 0, stream>>>(
      Wqkv, WqkvT, 3 * DIM_);
  transpose_w_kernel<<<dim3((DIM_ * DIM_) / 256), 256, 0, stream>>>(Wr, WrT, DIM_);
  transpose_w_kernel<<<dim3((DIM_ * DIM_) / 256), 256, 0, stream>>>(Wout, WoutT,
                                                                    DIM_);

  // 1) qkv projection: 256 M-tiles x 32 N-tiles, blockIdx.y = q/k/v
  qkv_proj_kernel<<<dim3((256 * 32) / 4, 3), 128, 0, stream>>>(
      xbf, WqkvT, u, v, qu, qvb, kk, vvT);
  // 2) relative-position projection: 128 x 32 tiles
  rel_proj_kernel<<<dim3((128 * 32) / 4), 128, 0, stream>>>(rbf, WrT, rh);
  // 3) fused attention: B*H*(T/16) = 2048 query tiles
  relattn_kernel<<<dim3((B_ * H_ * (T_ / 16)) / 4), 128, 0, stream>>>(
      qu, qvb, kk, vvT, rh, ao);
  // 4) output projection: 256 x 32 tiles
  out_proj_kernel<<<dim3((256 * 32) / 4), 128, 0, stream>>>(ao, WoutT, bout, out);
}
